// SequenceAKOrN_19439021981972
// MI455X (gfx1250) — compile-verified, matrix operand order checked by reading, not run on hardware
//
#include <hip/hip_runtime.h>
#include <hip/hip_bf16.h>
#include <math.h>

// ---------------------------------------------------------------------------
// AKOrN (attentive Kuramoto) for MI455X / gfx1250, wave32 + WMMA + TDM.
//
//  1. split x, Wq,Wk,Wv,Wo into bf16 hi/lo pairs (bf16x3 trick ~= fp32 GEMM)
//  2. q,k (bf16) and v (fp32) projections via v_wmma_f32_16x16x32_bf16,
//     tiles DMA'd into LDS by the Tensor Data Mover (double-buffered,
//     padding applied by TDM descriptor -> conflict-free 80B LDS rows)
//  3. phases = atan2(v0, v1+1e-8)
//  4. softmax stats (row max m, denom Z): online softmax over WMMA logits
//  5. 5x Kuramoto steps: flash-recompute logit tiles (J never materialized),
//     P = exp(l - m); accumulate P.[sin phi_j, cos phi_j] per lane + shfl tree;
//     coupling = (sin phi_i * Sc - cos phi_i * Ss)/Z; phi += DT*(omega+c) mod 2pi
//  6. vmod = v*cos(phi) split hi/lo, output GEMM + residual x -> d_out
// ---------------------------------------------------------------------------

typedef __bf16 bf16_t;
typedef __attribute__((ext_vector_type(16))) __bf16 v16bf;
typedef __attribute__((ext_vector_type(8)))  float  v8f;
typedef __attribute__((ext_vector_type(4)))  unsigned int v4u;
typedef __attribute__((ext_vector_type(8)))  int    v8i;
typedef __attribute__((ext_vector_type(4)))  int    v4i;

#define DIMD     1024
#define HEADS    16
#define DK       64
#define BATCH    2
#define SEQ      2048
#define NTOK     (BATCH * SEQ)   // 4096
#define DT_STEP  0.1f
#define NUM_STEPS 5
#define QK_SCALE 0.125f          // 1/sqrt(64)
#define TWO_PI_F 6.283185307179586f

// ---------------------------------------------------------------- helpers ---

__device__ inline v8f wmma_bf16(v16bf a, v16bf b, v8f c) {
  // (neg_a, A, neg_b, B, c_mod, C, reuse_a, reuse_b)
  return __builtin_amdgcn_wmma_f32_16x16x32_bf16(false, a, false, b,
                                                 (short)0, c, false, false);
}

__device__ inline v8f v8f_zero() {
  v8f z = {0.f, 0.f, 0.f, 0.f, 0.f, 0.f, 0.f, 0.f};
  return z;
}

__device__ inline unsigned short f2bf_bits(float f) {
  union { float f; unsigned u; } a; a.f = f;
  unsigned u = a.u;
  unsigned r = u + 0x7FFFu + ((u >> 16) & 1u);  // round-to-nearest-even
  return (unsigned short)(r >> 16);
}
__device__ inline bf16_t f2bf(float f) {
  unsigned short h = f2bf_bits(f);
  bf16_t b; __builtin_memcpy(&b, &h, 2); return b;
}
__device__ inline float bf2f(bf16_t b) {
  unsigned short h; __builtin_memcpy(&h, &b, 2);
  union { unsigned u; float f; } a; a.u = ((unsigned)h) << 16; return a.f;
}

// 32-byte fragment load assuming only 16B alignment (2x ds_read_b128)
__device__ inline v16bf load16(const bf16_t* p) {
  struct alignas(16) Pair { uint4 a, b; };
  Pair u;
  u.a = *(const uint4*)p;
  u.b = *(const uint4*)(p + 8);
  v16bf v; __builtin_memcpy(&v, &u, 32); return v;
}

// ------------------------------------------------ Tensor Data Mover --------
// 2D bf16 tile DMA: global (rows x row_elems, row stride in elements)
// -> LDS, optionally padded every 64 bytes with +16 bytes (pad_interval=3:
// 16 DWORDs, pad_amount=3: 4 DWORDs) so a 32-elem row lands on an 80B pitch.
// D# layout per CDNA5 ISA ch.8 (group0 128b, group1 256b; groups 2/3 unused
// for 2D tensors). Issue from ONE wave only: TDM ignores EXEC.

__device__ inline void tdm_load_2d_bf16(const void* gptr, unsigned lds_addr,
                                        unsigned rows, unsigned row_elems,
                                        unsigned row_stride_elems,
                                        bool pad_rows) {
  unsigned long long ga = (unsigned long long)(size_t)gptr;
  v4u g0;
  g0.x = 1u;                                     // count=1, user descriptor
  g0.y = lds_addr;                               // lds_addr (bytes)
  g0.z = (unsigned)(ga & 0xFFFFFFFFu);           // global_addr[31:0]
  g0.w = (unsigned)((ga >> 32) & 0x1FFFFFFu)     // global_addr[56:32]
       | (2u << 30);                             // type = 2 ("image")

  // group1 word0: data_size=1 (2-byte elems) @ [17:16]; pad_enable @ [20];
  // pad_interval @ [24:22] (3 -> every 16 DWORDs = 64B);
  // pad_amount @ [31:25] (3 -> 4 DWORDs = 16B)
  unsigned w0 = (1u << 16);
  if (pad_rows) w0 |= (1u << 20) | (3u << 22) | (3u << 25);
  unsigned long long td0 = row_elems;            // tensor_dim0 (elems)
  unsigned long long td1 = rows;                 // tensor_dim1
  unsigned long long st0 = row_stride_elems;     // tensor_dim0_stride (48b)
  unsigned w1 = (unsigned)((td0 & 0xFFFFu) << 16);                     // td0 lo
  unsigned w2 = (unsigned)((td0 >> 16) & 0xFFFFu)
              | (unsigned)((td1 & 0xFFFFu) << 16);                     // td1 lo
  unsigned w3 = (unsigned)((td1 >> 16) & 0xFFFFu)
              | ((row_elems & 0xFFFFu) << 16);                         // tile_dim0
  unsigned w4 = (rows & 0xFFFFu);                // tile_dim1 (tile_dim2 = 0)
  unsigned w5 = (unsigned)(st0 & 0xFFFFFFFFu);
  unsigned w6 = (unsigned)((st0 >> 32) & 0xFFFFu);   // dim1_stride = 0
  unsigned w7 = 0u;
  v8i g1 = {(int)w0, (int)w1, (int)w2, (int)w3,
            (int)w4, (int)w5, (int)w6, (int)w7};
  v4i z4 = {0, 0, 0, 0};
  v8i z8 = {0, 0, 0, 0, 0, 0, 0, 0};
  // clang-23 / therock-10.0 arity: (g0, g1, g2, g3, g4, cpol)
  __builtin_amdgcn_tensor_load_to_lds(g0, g1, z4, z4, z8, 0);
}

__device__ inline unsigned lds_off(const void* p) {
  return (unsigned)(size_t)p;  // flat LDS addr low 32 bits = LDS byte offset
}

// ---------------------------------------------------- fp32 -> bf16 hi/lo ---

__global__ void split_bf16x2_kernel(const float* __restrict__ src,
                                    bf16_t* __restrict__ hi,
                                    bf16_t* __restrict__ lo, int n) {
  int i = blockIdx.x * blockDim.x + threadIdx.x;
  if (i >= n) return;
  float f = src[i];
  bf16_t h = f2bf(f);
  hi[i] = h;
  lo[i] = f2bf(f - bf2f(h));
}

// -------------------------------------------------- bf16x3 WMMA GEMM -------
// C(MxN) = A(MxK) * W(NxK)^T. Block tile 64M x 128N, 8 waves (4xM, 2xN),
// K-step 32. Tiles double-buffered via TDM; WMMA overlaps next-slab DMA.

template <int OUT_BF16>
__global__ __launch_bounds__(256)
void gemm_bf16x3_kernel(const bf16_t* __restrict__ Ahi,
                        const bf16_t* __restrict__ Alo,
                        const bf16_t* __restrict__ Bhi,
                        const bf16_t* __restrict__ Blo,
                        float* __restrict__ Cf, bf16_t* __restrict__ Cb,
                        const float* __restrict__ residual,
                        int M, int N, int K) {
  // TDM pads each 64B (32-elem) row by 16B -> pitch 40 elems (80B).
  __shared__ __align__(16) bf16_t sAh[2][64][40];
  __shared__ __align__(16) bf16_t sAl[2][64][40];
  __shared__ __align__(16) bf16_t sBh[2][128][40];
  __shared__ __align__(16) bf16_t sBl[2][128][40];

  const int t    = threadIdx.x;
  const int lane = t & 31;
  const int wave = t >> 5;
  const int wm   = wave & 3;
  const int wn   = wave >> 2;
  const int m0   = blockIdx.x * 64;
  const int n0   = blockIdx.y * 128;
  const int mBase = wm * 16;
  const int nBase = wn * 64;
  const int khalf = (lane >> 4) * 16;

  v8f acc[4];
  for (int i = 0; i < 4; ++i) acc[i] = v8f_zero();

  auto issue_slab = [&](int k0, int buf) {
    tdm_load_2d_bf16(&Ahi[(size_t)m0 * K + k0], lds_off(&sAh[buf][0][0]),
                     64, 32, (unsigned)K, true);
    tdm_load_2d_bf16(&Alo[(size_t)m0 * K + k0], lds_off(&sAl[buf][0][0]),
                     64, 32, (unsigned)K, true);
    tdm_load_2d_bf16(&Bhi[(size_t)n0 * K + k0], lds_off(&sBh[buf][0][0]),
                     128, 32, (unsigned)K, true);
    tdm_load_2d_bf16(&Blo[(size_t)n0 * K + k0], lds_off(&sBl[buf][0][0]),
                     128, 32, (unsigned)K, true);
  };

  if (wave == 0) issue_slab(0, 0);

  const int nSlabs = K / 32;
  for (int c = 0; c < nSlabs; ++c) {
    const int cur = c & 1;
    const int nxt = cur ^ 1;
    __syncthreads();  // buf[nxt] no longer being read (from slab c-1)
    if (wave == 0) {
      if (c + 1 < nSlabs) {
        issue_slab((c + 1) * 32, nxt);
        __builtin_amdgcn_s_wait_tensorcnt(4);  // slab c done (in-order)
      } else {
        __builtin_amdgcn_s_wait_tensorcnt(0);
      }
    }
    __syncthreads();  // slab c visible to all waves

    v16bf ah = load16(&sAh[cur][mBase + (lane & 15)][khalf]);
    v16bf al = load16(&sAl[cur][mBase + (lane & 15)][khalf]);
    for (int nt = 0; nt < 4; ++nt) {
      v16bf bh = load16(&sBh[cur][nBase + nt * 16 + (lane & 15)][khalf]);
      v16bf bl = load16(&sBl[cur][nBase + nt * 16 + (lane & 15)][khalf]);
      acc[nt] = wmma_bf16(ah, bl, acc[nt]);  // hi*lo
      acc[nt] = wmma_bf16(al, bh, acc[nt]);  // lo*hi
      acc[nt] = wmma_bf16(ah, bh, acc[nt]);  // hi*hi
    }
  }

  // C frag layout: VGPR r, lanes 0-15 -> row r, lanes 16-31 -> row r+8
  const int rowOff = m0 + mBase + ((lane >> 4) * 8);
  for (int nt = 0; nt < 4; ++nt) {
    int col = n0 + nBase + nt * 16 + (lane & 15);
    for (int r = 0; r < 8; ++r) {
      int row = rowOff + r;
      float val = acc[nt][r];
      if (OUT_BF16) {
        Cb[(size_t)row * N + col] = f2bf(val);
      } else {
        if (residual) val += residual[(size_t)row * N + col];
        Cf[(size_t)row * N + col] = val;
      }
    }
  }
}

// ---------------------------------------------------------- phase init -----

__global__ void phase_init_kernel(const float* __restrict__ v,
                                  float* __restrict__ phases) {
  int i = blockIdx.x * blockDim.x + threadIdx.x;  // over B*H*T
  if (i >= BATCH * HEADS * SEQ) return;
  int tpos = i % SEQ;
  int bh   = i / SEQ;
  int h    = bh % HEADS;
  int b    = bh / HEADS;
  size_t base = ((size_t)(b * SEQ + tpos)) * DIMD + (size_t)h * DK;
  phases[i] = atan2f(v[base], v[base + 1] + 1e-8f);
}

// ---------------------------------------------- softmax stats (m, Z) -------
// One block = 128 query rows of one (b,h); 8 waves, each owns a 16-row tile.
// K tiles (128 keys x 64 dims) double-buffered via TDM.

__global__ __launch_bounds__(256)
void softmax_stats_kernel(const bf16_t* __restrict__ q,
                          const bf16_t* __restrict__ k,
                          float* __restrict__ mOut, float* __restrict__ zOut) {
  __shared__ __align__(16) bf16_t sK[2][128][64];
  const int t    = threadIdx.x;
  const int lane = t & 31;
  const int wave = t >> 5;
  const int bh   = blockIdx.y;
  const int b    = bh / HEADS;
  const int h    = bh % HEADS;
  const int tq0  = blockIdx.x * 128 + wave * 16;
  const size_t hdBase = (size_t)b * SEQ * DIMD + (size_t)h * DK;
  const int khalf = (lane >> 4) * 16;

  const bf16_t* qrow = q + hdBase + (size_t)(tq0 + (lane & 15)) * DIMD;
  v16bf qa0 = load16(qrow + khalf);        // K = 0..31
  v16bf qa1 = load16(qrow + 32 + khalf);   // K = 32..63

  float mx[8], sm[8];
  for (int r = 0; r < 8; ++r) { mx[r] = -1e30f; sm[r] = 0.f; }

  if (wave == 0)
    tdm_load_2d_bf16(k + hdBase, lds_off(&sK[0][0][0]), 128, 64, DIMD, false);

  const int nch = SEQ / 128;
  for (int c = 0; c < nch; ++c) {
    const int cur = c & 1;
    const int nxt = cur ^ 1;
    __syncthreads();
    if (wave == 0) {
      if (c + 1 < nch) {
        tdm_load_2d_bf16(k + hdBase + (size_t)(c + 1) * 128 * DIMD,
                         lds_off(&sK[nxt][0][0]), 128, 64, DIMD, false);
        __builtin_amdgcn_s_wait_tensorcnt(1);
      } else {
        __builtin_amdgcn_s_wait_tensorcnt(0);
      }
    }
    __syncthreads();
    for (int nt = 0; nt < 8; ++nt) {
      const bf16_t* kr = &sK[cur][nt * 16 + (lane & 15)][0];
      v16bf b0 = load16(kr + khalf);
      v16bf b1 = load16(kr + 32 + khalf);
      v8f acc = v8f_zero();
      acc = wmma_bf16(qa0, b0, acc);
      acc = wmma_bf16(qa1, b1, acc);
      for (int r = 0; r < 8; ++r) {
        float l  = acc[r] * QK_SCALE;
        float nm = fmaxf(mx[r], l);
        sm[r] = sm[r] * __expf(mx[r] - nm) + __expf(l - nm);
        mx[r] = nm;
      }
    }
  }
  for (int off = 1; off < 16; off <<= 1) {
    for (int r = 0; r < 8; ++r) {
      float om = __shfl_xor(mx[r], off);
      float os = __shfl_xor(sm[r], off);
      float nm = fmaxf(mx[r], om);
      sm[r] = sm[r] * __expf(mx[r] - nm) + os * __expf(om - nm);
      mx[r] = nm;
    }
  }
  int rr = lane & 15;
  if (rr < 8) {
    int row = tq0 + rr + 8 * (lane >> 4);
    mOut[(size_t)bh * SEQ + row] = mx[rr];
    zOut[(size_t)bh * SEQ + row] = sm[rr];
  }
}

// --------------------------------------------------- Kuramoto step ---------
// coupling_i = sin(phi_i)*Sum_j J_ij cos(phi_j) - cos(phi_i)*Sum_j J_ij sin(phi_j)
// Flash-recompute logit tiles with WMMA; K tiles via double-buffered TDM.

__global__ __launch_bounds__(256)
void kuramoto_step_kernel(const bf16_t* __restrict__ q,
                          const bf16_t* __restrict__ k,
                          const float* __restrict__ phIn,
                          float* __restrict__ phOut,
                          const float* __restrict__ mIn,
                          const float* __restrict__ zIn,
                          const float* __restrict__ omega) {
  __shared__ __align__(16) bf16_t sK[2][128][64];
  __shared__ float sSin[128];
  __shared__ float sCos[128];
  const int t    = threadIdx.x;
  const int lane = t & 31;
  const int wave = t >> 5;
  const int bh   = blockIdx.y;
  const int b    = bh / HEADS;
  const int h    = bh % HEADS;
  const int tq0  = blockIdx.x * 128 + wave * 16;
  const size_t hdBase = (size_t)b * SEQ * DIMD + (size_t)h * DK;
  const size_t phBase = (size_t)bh * SEQ;
  const int khalf = (lane >> 4) * 16;

  const bf16_t* qrow = q + hdBase + (size_t)(tq0 + (lane & 15)) * DIMD;
  v16bf qa0 = load16(qrow + khalf);
  v16bf qa1 = load16(qrow + 32 + khalf);

  float mrow[8];
  for (int r = 0; r < 8; ++r)
    mrow[r] = mIn[phBase + tq0 + r + 8 * (lane >> 4)];

  float sAcc[8], cAcc[8];
  for (int r = 0; r < 8; ++r) { sAcc[r] = 0.f; cAcc[r] = 0.f; }

  if (wave == 0)
    tdm_load_2d_bf16(k + hdBase, lds_off(&sK[0][0][0]), 128, 64, DIMD, false);

  const int nch = SEQ / 128;
  for (int c = 0; c < nch; ++c) {
    const int cur = c & 1;
    const int nxt = cur ^ 1;
    __syncthreads();
    if (t < 128) {
      float p = phIn[phBase + c * 128 + t];
      sSin[t] = __sinf(p);
      sCos[t] = __cosf(p);
    }
    if (wave == 0) {
      if (c + 1 < nch) {
        tdm_load_2d_bf16(k + hdBase + (size_t)(c + 1) * 128 * DIMD,
                         lds_off(&sK[nxt][0][0]), 128, 64, DIMD, false);
        __builtin_amdgcn_s_wait_tensorcnt(1);
      } else {
        __builtin_amdgcn_s_wait_tensorcnt(0);
      }
    }
    __syncthreads();
    for (int nt = 0; nt < 8; ++nt) {
      const bf16_t* kr = &sK[cur][nt * 16 + (lane & 15)][0];
      v16bf b0 = load16(kr + khalf);
      v16bf b1 = load16(kr + 32 + khalf);
      v8f acc = v8f_zero();
      acc = wmma_bf16(qa0, b0, acc);
      acc = wmma_bf16(qa1, b1, acc);
      int   jl = nt * 16 + (lane & 15);
      float sj = sSin[jl];
      float cj = sCos[jl];
      for (int r = 0; r < 8; ++r) {
        float p = __expf(acc[r] * QK_SCALE - mrow[r]);  // unnormalized J
        sAcc[r] += p * sj;
        cAcc[r] += p * cj;
      }
    }
  }
  for (int off = 1; off < 16; off <<= 1) {
    for (int r = 0; r < 8; ++r) {
      sAcc[r] += __shfl_xor(sAcc[r], off);
      cAcc[r] += __shfl_xor(cAcc[r], off);
    }
  }
  int rr = lane & 15;
  if (rr < 8) {
    int row = tq0 + rr + 8 * (lane >> 4);
    float phi = phIn[phBase + row];
    float z   = zIn[phBase + row];
    float coupling = (__sinf(phi) * cAcc[rr] - __cosf(phi) * sAcc[rr]) / z;
    float np = phi + DT_STEP * (omega[h] + coupling);
    np = fmodf(np, TWO_PI_F);
    if (np < 0.f) np += TWO_PI_F;
    phOut[phBase + row] = np;
  }
}

// ------------------------------------------------ v * cos(phi), split ------

__global__ void vmod_kernel(const float* __restrict__ v,
                            const float* __restrict__ phases,
                            bf16_t* __restrict__ hi, bf16_t* __restrict__ lo) {
  int i = blockIdx.x * blockDim.x + threadIdx.x;  // over NTOK*DIMD
  if (i >= NTOK * DIMD) return;
  int d    = i % DIMD;
  int tok  = i / DIMD;
  int h    = d / DK;
  int b    = tok / SEQ;
  int tpos = tok % SEQ;
  float c   = __cosf(phases[((size_t)(b * HEADS + h)) * SEQ + tpos]);
  float val = v[i] * c;
  bf16_t hh = f2bf(val);
  hi[i] = hh;
  lo[i] = f2bf(val - bf2f(hh));
}

// ----------------------------------------------------------- launcher ------

extern "C" void kernel_launch(void* const* d_in, const int* in_sizes, int n_in,
                              void* d_out, int out_size, void* d_ws,
                              size_t ws_size, hipStream_t stream) {
  (void)in_sizes; (void)n_in; (void)out_size; (void)ws_size;
  const float* x     = (const float*)d_in[0];
  const float* Wq    = (const float*)d_in[1];
  const float* Wk    = (const float*)d_in[2];
  const float* Wv    = (const float*)d_in[3];
  const float* Wo    = (const float*)d_in[4];
  const float* omega = (const float*)d_in[5];

  char* p = (char*)d_ws;
  auto carve = [&](size_t bytes) -> void* {
    void* r = (void*)p;
    p += (bytes + 255) & ~(size_t)255;
    return r;
  };
  const size_t XE = (size_t)NTOK * DIMD;          // 4,194,304
  const size_t WE = (size_t)DIMD * DIMD;          // 1,048,576
  const size_t PE = (size_t)BATCH * HEADS * SEQ;  // 65,536

  bf16_t* xhi = (bf16_t*)carve(XE * 2);
  bf16_t* xlo = (bf16_t*)carve(XE * 2);
  bf16_t* wqh = (bf16_t*)carve(WE * 2);
  bf16_t* wql = (bf16_t*)carve(WE * 2);
  bf16_t* wkh = (bf16_t*)carve(WE * 2);
  bf16_t* wkl = (bf16_t*)carve(WE * 2);
  bf16_t* wvh = (bf16_t*)carve(WE * 2);
  bf16_t* wvl = (bf16_t*)carve(WE * 2);
  bf16_t* woh = (bf16_t*)carve(WE * 2);
  bf16_t* wol = (bf16_t*)carve(WE * 2);
  bf16_t* qb  = (bf16_t*)carve(XE * 2);
  bf16_t* kb  = (bf16_t*)carve(XE * 2);
  float*  vf  = (float*)carve(XE * 4);
  bf16_t* vmh = (bf16_t*)carve(XE * 2);
  bf16_t* vml = (bf16_t*)carve(XE * 2);
  float*  ph0 = (float*)carve(PE * 4);
  float*  ph1 = (float*)carve(PE * 4);
  float*  mB  = (float*)carve(PE * 4);
  float*  zB  = (float*)carve(PE * 4);

  split_bf16x2_kernel<<<(int)((XE + 255) / 256), 256, 0, stream>>>(x, xhi, xlo, (int)XE);
  split_bf16x2_kernel<<<(int)((WE + 255) / 256), 256, 0, stream>>>(Wq, wqh, wql, (int)WE);
  split_bf16x2_kernel<<<(int)((WE + 255) / 256), 256, 0, stream>>>(Wk, wkh, wkl, (int)WE);
  split_bf16x2_kernel<<<(int)((WE + 255) / 256), 256, 0, stream>>>(Wv, wvh, wvl, (int)WE);
  split_bf16x2_kernel<<<(int)((WE + 255) / 256), 256, 0, stream>>>(Wo, woh, wol, (int)WE);

  dim3 gemmGrid(NTOK / 64, DIMD / 128);
  gemm_bf16x3_kernel<1><<<gemmGrid, 256, 0, stream>>>(
      xhi, xlo, wqh, wql, nullptr, qb, nullptr, NTOK, DIMD, DIMD);
  gemm_bf16x3_kernel<1><<<gemmGrid, 256, 0, stream>>>(
      xhi, xlo, wkh, wkl, nullptr, kb, nullptr, NTOK, DIMD, DIMD);
  gemm_bf16x3_kernel<0><<<gemmGrid, 256, 0, stream>>>(
      xhi, xlo, wvh, wvl, vf, nullptr, nullptr, NTOK, DIMD, DIMD);

  phase_init_kernel<<<(int)((PE + 255) / 256), 256, 0, stream>>>(vf, ph0);

  dim3 attnGrid(SEQ / 128, BATCH * HEADS);
  softmax_stats_kernel<<<attnGrid, 256, 0, stream>>>(qb, kb, mB, zB);

  float* pin = ph0;
  float* pout = ph1;
  for (int s = 0; s < NUM_STEPS; ++s) {
    kuramoto_step_kernel<<<attnGrid, 256, 0, stream>>>(qb, kb, pin, pout, mB, zB, omega);
    float* tmp = pin; pin = pout; pout = tmp;
  }
  // final phases now in `pin`

  vmod_kernel<<<(int)((XE + 255) / 256), 256, 0, stream>>>(vf, pin, vmh, vml);

  gemm_bf16x3_kernel<0><<<gemmGrid, 256, 0, stream>>>(
      vmh, vml, woh, wol, (float*)d_out, nullptr, x, NTOK, DIMD, DIMD);
}
